// PE_LSTM_22239340658892
// MI455X (gfx1250) — matrix operand order repaired and checked by reading, not compiled, not run on hardware
//
#include <hip/hip_runtime.h>

// B=32, T=2048, D=H=256, 4H=1024
#define T_STEPS 2048
#define HDIM    256
#define G4      1024   // 4*H
#define KTOT    512    // D + H (fused [x_t, h] @ [W; R])
#define NTILES  64     // G4 / 16
#define KTILES  16     // KTOT / 32

typedef __attribute__((ext_vector_type(16))) _Float16 v16h;
typedef __attribute__((ext_vector_type(8)))  _Float16 v8h;
typedef __attribute__((ext_vector_type(8)))  float    v8f;

// ---------------------------------------------------------------------------
// Prep: convert kernel[256,1024] (f32) + recurrent_kernel[256,1024] (f32) into
// a single fp16 stacked matrix Wcat[512,1024], laid out in WMMA-B-operand
// tile-blocked order: block (nt, kt) is 1 KB; lane l's 16 fp16 values are
// contiguous 32 B:  n = nt*16 + (l%16),  k = kt*32 + (l<16 ? 0 : 16) + e.
// Total: 64*16*32*32 B = 1 MB in d_ws (stays hot in the 192 MB L2).
// ---------------------------------------------------------------------------
__global__ void prep_weights(const float* __restrict__ kern,
                             const float* __restrict__ rk,
                             _Float16* __restrict__ wb)
{
    int idx  = blockIdx.x * blockDim.x + threadIdx.x;   // 0 .. 32767
    int lane = idx & 31;
    int kt   = (idx >> 5) & 15;
    int nt   = idx >> 9;                                 // 0 .. 63
    int kbase = kt * 32 + ((lane & 16) ? 16 : 0);
    int n     = (nt << 4) + (lane & 15);
    v16h v;
#pragma unroll
    for (int e = 0; e < 16; ++e) {
        int k = kbase + e;
        float f = (k < HDIM) ? kern[(size_t)k * G4 + n]
                             : rk[(size_t)(k - HDIM) * G4 + n];
        v[e] = (_Float16)f;
    }
    *(v16h*)(wb + (size_t)idx * 16) = v;
}

// ---------------------------------------------------------------------------
// Persistent recurrent kernel: 16 workgroups, 2 batch rows each, 1024 threads
// (32 waves). Per step: z = [x_t, h] @ Wcat via v_wmma_f32_16x16x32_f16
// (2 N-tiles per wave, 16 K-tiles, unroll 8: deep pipeline without tripping
// LICM into spilling B tiles across the t-loop), then gates on 512 threads
// while the other 512 prefetch x_{t+1}. A-operand LDS buffer is ping-pong
// double-buffered.
// ---------------------------------------------------------------------------
__global__ __launch_bounds__(1024) void lstm_recurrent(
    const float* __restrict__ x,         // [32, 2048, 256]
    const _Float16* __restrict__ wb,     // blocked Wcat (1 MB, in ws)
    const float* __restrict__ bias,      // [1024]
    float* __restrict__ out)             // [32, 2048, 256]
{
    // A buffers: 16 rows (2 valid) x 512 fp16 = 16 KB each, double buffered.
    __shared__ __align__(32) unsigned char abuf[2][16 * 1024];
    __shared__ float z_lds[2][G4];     // 8 KB
    __shared__ float c_lds[2][HDIM];   // 2 KB
    __shared__ float bias_lds[G4];     // 4 KB

    const int tid  = threadIdx.x;
    const int lane = tid & 31;
    const int wave = tid >> 5;
    const int b0   = blockIdx.x * 2;   // this WG owns batch rows b0, b0+1

    // ---- init: h0 = 0, c0 = 0, bias into LDS, prefetch x_0 ----
    if (tid < 512) {
        int r = tid >> 8, j = tid & 255;
        c_lds[r][j] = 0.0f;
        *(_Float16*)(&abuf[0][r * 1024 + 512 + 2 * j]) = (_Float16)0.0f;  // h part
    } else {
        int t2 = tid - 512;
        int r = t2 >> 8, j = t2 & 255;
        float xv = x[((size_t)(b0 + r) * T_STEPS + 0) * HDIM + j];
        *(_Float16*)(&abuf[0][r * 1024 + 2 * j]) = (_Float16)xv;          // x part
    }
    bias_lds[tid] = bias[tid];
    __syncthreads();

    // A-operand per-lane addressing (this code's contiguous k order, same as Wcat B tiles):
// lane<16 : row=lane,    K {0..15}
// lane>=16: row=lane-16, K {16..31}
    const int arow  = lane & 15;
    const int ahalf = (lane >> 4) & 1;
    const int nt0   = wave * 2;
    const int nt1   = wave * 2 + 1;

    // Hoisted B-operand base pointers: per-K-tile stride is a constant 1 KB,
    // so loads become immediate-offset b128 loads from a shared base.
    const _Float16* bptr0 = wb + ((size_t)(nt0 * KTILES) * 32 + lane) * 16;
    const _Float16* bptr1 = wb + ((size_t)(nt1 * KTILES) * 32 + lane) * 16;
    const int abase0 = arow * 1024 + ahalf * 32;

    for (int t = 0; t < T_STEPS; ++t) {
        const unsigned char* acur = abuf[t & 1];
        v8f acc0 = {};
        v8f acc1 = {};
#pragma unroll 8
        for (int kt = 0; kt < KTILES; ++kt) {
            v8h alo = *(const v8h*)(acur + abase0 + kt * 64);        // K chunk 0
            v8h ahi = *(const v8h*)(acur + abase0 + kt * 64 + 16);   // K chunk 1
            v16h a;
#pragma unroll
            for (int e = 0; e < 8; ++e) { a[e] = alo[e]; a[8 + e] = ahi[e]; }

            v16h bm0 = *(const v16h*)(bptr0 + kt * 512);   // +kt KB, immediate offset
            v16h bm1 = *(const v16h*)(bptr1 + kt * 512);

            acc0 = __builtin_amdgcn_wmma_f32_16x16x32_f16(
                       false, a, false, bm0, (short)0, acc0, false, false);
            acc1 = __builtin_amdgcn_wmma_f32_16x16x32_f16(
                       false, a, false, bm1, (short)0, acc1, false, false);
        }

        // C/D layout: VGPR v, lanes 0-15 -> (M=v, N=lane). Rows 0,1 are the
        // two valid batch rows -> acc[0], acc[1] on lanes 0-15.
        if (lane < 16) {
            z_lds[0][nt0 * 16 + lane] = acc0[0];
            z_lds[1][nt0 * 16 + lane] = acc0[1];
            z_lds[0][nt1 * 16 + lane] = acc1[0];
            z_lds[1][nt1 * 16 + lane] = acc1[1];
        }
        __syncthreads();

        unsigned char* anext = (unsigned char*)abuf[(t + 1) & 1];
        if (tid < 512) {
            // gate phase: one (row, h-column) per thread
            int r = tid >> 8, j = tid & 255;
            float zi = z_lds[r][j]           + bias_lds[j];
            float zf = z_lds[r][HDIM + j]    + bias_lds[HDIM + j];
            float zg = z_lds[r][2*HDIM + j]  + bias_lds[2*HDIM + j];
            float zo = z_lds[r][3*HDIM + j]  + bias_lds[3*HDIM + j];
            float ig = 1.0f / (1.0f + __expf(-zi));
            float fg = 1.0f / (1.0f + __expf(-zf));
            float gg = zg > 0.0f ? zg : 0.0f;            // relu candidate
            float og = 1.0f / (1.0f + __expf(-zo));
            float c  = fg * c_lds[r][j] + ig * gg;
            c_lds[r][j] = c;
            float h = og * (c > 0.0f ? c : 0.0f);        // relu on cell output
            // stage h_{t+1} operand (fp16) into next A buffer
            *(_Float16*)(anext + r * 1024 + 512 + 2 * j) = (_Float16)h;
            // residual: out = x + h
            size_t xo = ((size_t)(b0 + r) * T_STEPS + t) * HDIM + j;
            out[xo] = x[xo] + h;
        } else {
            // prefetch x_{t+1} into next A buffer (fp16)
            int t2 = tid - 512;
            int r = t2 >> 8, j = t2 & 255;
            if (t + 1 < T_STEPS) {
                float xv = x[((size_t)(b0 + r) * T_STEPS + (t + 1)) * HDIM + j];
                *(_Float16*)(anext + r * 1024 + 2 * j) = (_Float16)xv;
            }
        }
        __syncthreads();
    }
}

// ---------------------------------------------------------------------------
extern "C" void kernel_launch(void* const* d_in, const int* in_sizes, int n_in,
                              void* d_out, int out_size, void* d_ws, size_t ws_size,
                              hipStream_t stream)
{
    const float* x    = (const float*)d_in[0];  // [32,2048,256]
    const float* kern = (const float*)d_in[1];  // [256,1024]
    const float* rk   = (const float*)d_in[2];  // [256,1024]
    const float* bias = (const float*)d_in[3];  // [1024]
    float* out = (float*)d_out;

    _Float16* wb = (_Float16*)d_ws;  // needs 1 MB: 64*16*32 lanes * 32 B

    // 64 ntiles * 16 ktiles * 32 lanes = 32768 threads
    prep_weights<<<128, 256, 0, stream>>>(kern, rk, wb);
    lstm_recurrent<<<16, 1024, 0, stream>>>(x, wb, bias, out);
}
